// MultiHeadSelfAttention_30545807409622
// MI455X (gfx1250) — matrix operand order repaired and checked by reading, not compiled, not run on hardware
//
#include <hip/hip_runtime.h>

// MultiHeadSelfAttention: B=4, S=2048, D=128, H=8
#define Bv 4
#define Sv 2048
#define Dv 128
#define Hv 8

typedef __attribute__((ext_vector_type(16))) __bf16    v16bf;
typedef __attribute__((ext_vector_type(8)))  float     v8f;
typedef __attribute__((ext_vector_type(4)))  unsigned  u32x4;

#define NEG_INF (-__builtin_inff())

// One WMMA operand fragment: 16 bf16 values = 8 VGPRs (wave32).
union Frag {
  v16bf  v;
  __bf16 e[16];
  u32x4  q[2];
};

__device__ __forceinline__ v8f wmma_bf16(const Frag& a, const Frag& b, v8f c) {
  return __builtin_amdgcn_wmma_f32_16x16x32_bf16(
      /*neg_a=*/false, a.v, /*neg_b=*/false, b.v,
      /*c_mod=*/(short)0, c, /*reuse_a=*/false, /*reuse_b=*/false);
}

// A-matrix fragment (16x32 bf16, row-major source).
// ISA layout: lanes 0-15 hold K={kb..kb+7, kb+16..kb+23}, lanes 16-31 hold
// K={kb+8..kb+15, kb+24..kb+31}.  Caller passes p = row_base + kb + (lo?0:8);
// elements [0..7] at p, [8..15] at p+16.
__device__ __forceinline__ Frag load_frag_a(const __bf16* p) {
  Frag f;
  f.q[0] = *(const u32x4*)(p);
  f.q[1] = *(const u32x4*)(p + 16);
  return f;
}

// B-matrix fragment (32x16 bf16): lane holds column N=lane%16, same K order as A
// (lanes 0-15: kb..kb+7, kb+16..kb+23; lanes 16-31: kb+8..kb+15, kb+24..kb+31).
// Source must be column-major (K contiguous): p = col_base + kb + (lo?0:8).
__device__ __forceinline__ Frag load_frag_b(const __bf16* p) {
  Frag f;
  f.q[0] = *(const u32x4*)(p);
  f.q[1] = *(const u32x4*)(p + 16);
  return f;
}

// Async DMA: one 16B chunk global -> LDS, tracked by ASYNCcnt.
__device__ __forceinline__ void async_b128(unsigned lds_addr, const __bf16* g) {
  asm volatile("global_load_async_to_lds_b128 %0, %1, off"
               :: "v"(lds_addr), "v"((unsigned long long)(uintptr_t)g)
               : "memory");
}
__device__ __forceinline__ void wait_async0() {
  asm volatile("s_wait_asynccnt 0" ::: "memory");
}

// ---------------- prep kernels ----------------
__global__ void cvt_bf16(const float* __restrict__ s, __bf16* __restrict__ d, int n) {
  int i = blockIdx.x * blockDim.x + threadIdx.x;
  if (i < n) d[i] = (__bf16)s[i];
}

// d[c*rows + r] = s[r*cols + c]  (transpose + convert, so B-fragments are contiguous)
__global__ void transpose_bf16(const float* __restrict__ s, __bf16* __restrict__ d,
                               int rows, int cols) {
  int i = blockIdx.x * blockDim.x + threadIdx.x;
  if (i < rows * cols) {
    int r = i / cols, c = i % cols;
    d[(size_t)c * rows + r] = (__bf16)s[i];
  }
}

// ---------------- QKV projection: (8192x128) @ (128x1024) x3 ----------------
// grid = (64, 24); 8 waves/block, wave = 16 rows x 128 cols, K-loop 4x32.
// Q,K stored head-major (B,H,S,D); V stored head-major TRANSPOSED (B,H,D,S)
// so the attention kernel can stage V tiles with straight async DMA copies.
__global__ __launch_bounds__(256) void qkv_gemm(
    const __bf16* __restrict__ xb,
    const __bf16* __restrict__ wqT, const __bf16* __restrict__ wkT,
    const __bf16* __restrict__ wvT,
    const float* __restrict__ bq, const float* __restrict__ bk,
    const float* __restrict__ bv,
    __bf16* __restrict__ qo, __bf16* __restrict__ ko, __bf16* __restrict__ vTo) {
  const int lane = threadIdx.x & 31;
  const int wave = threadIdx.x >> 5;
  const bool lo  = lane < 16;
  const int ln16 = lane & 15;

  const int jblk = blockIdx.y * 128;      // 0..3071 across {Wq,Wk,Wv}
  const int mat  = jblk >> 10;
  const int ncol = jblk & 1023;
  const __bf16* wT  = (mat == 0) ? wqT : (mat == 1) ? wkT : wvT;
  const float* bias = (mat == 0) ? bq  : (mat == 1) ? bk  : bv;
  __bf16* dst       = (mat == 0) ? qo  : (mat == 1) ? ko  : vTo;

  const int m0   = blockIdx.x * 128 + wave * 16;
  const int arow = m0 + ln16;

  v8f acc[8] = {};
#pragma unroll
  for (int kb = 0; kb < 128; kb += 32) {
    Frag a = load_frag_a(xb + (size_t)arow * 128 + kb + (lo ? 0 : 8));
#pragma unroll
    for (int n = 0; n < 8; ++n) {
      const int col = ncol + n * 16 + ln16;
      Frag bf = load_frag_b(wT + (size_t)col * 128 + kb + (lo ? 0 : 8));
      acc[n] = wmma_bf16(a, bf, acc[n]);
    }
  }
#pragma unroll
  for (int n = 0; n < 8; ++n) {
    const int j  = ncol + n * 16 + ln16;
    const float bval = bias[j];
    const int h = j >> 7, d = j & 127;
#pragma unroll
    for (int r = 0; r < 8; ++r) {
      const int row = m0 + r + (lo ? 0 : 8);
      const int bat = row >> 11, s = row & 2047;
      const __bf16 val = (__bf16)(acc[n][r] + bval);
      if (mat == 2)  // V transposed: [bh][d][s]
        dst[(((size_t)bat * Hv + h) * Dv + d) * Sv + s] = val;
      else           // Q,K: [bh][s][d]
        dst[(((size_t)bat * Hv + h) * Sv + s) * Dv + d] = val;
    }
  }
}

// ---------------- flash attention ----------------
// grid = (S/128, B*H); 8 waves/block, wave owns 16 query rows.
// 64-key tiles; K and V^T staged into double-buffered LDS with async DMA
// (issue next tile, compute current, s_wait_asynccnt 0 + barrier).
__global__ __launch_bounds__(256) void flash_attn(
    const __bf16* __restrict__ qb, const __bf16* __restrict__ kbp,
    const __bf16* __restrict__ vTb, const unsigned char* __restrict__ mask,
    __bf16* __restrict__ agg) {
  __shared__ __align__(16) __bf16 KsBuf[2][64 * 128];   // K tile [key][d]
  __shared__ __align__(16) __bf16 VtBuf[2][128 * 64];   // V^T tile [d][key]
  __shared__ __align__(16) __bf16 Ps[8][16 * 64];       // per-wave P tile

  const int lane = threadIdx.x & 31;
  const int wave = threadIdx.x >> 5;
  const bool lo  = lane < 16;
  const int ln16 = lane & 15;
  const int t    = threadIdx.x;

  const int bh = blockIdx.y;
  const int b  = bh >> 3;
  const int h  = bh & 7;
  const size_t head = (size_t)bh * Sv * Dv;
  const __bf16* Q  = qb  + head;
  const __bf16* K  = kbp + head;   // [s][d]
  const __bf16* vT = vTb + head;   // [d][s]

  const int m0 = blockIdx.x * 128 + wave * 16;

  // Q fragments for this wave's 16 rows (full d=128, 4 K-slices of 32)
  Frag aq[4];
#pragma unroll
  for (int kk = 0; kk < 4; ++kk)
    aq[kk] = load_frag_a(Q + (size_t)(m0 + ln16) * 128 + kk * 32 + (lo ? 0 : 8));

  v8f acc[8] = {};
  float mrow[8], lrow[8];
#pragma unroll
  for (int r = 0; r < 8; ++r) { mrow[r] = NEG_INF; lrow[r] = 0.f; }

  // async stage of tile kt into one LDS buffer (all 256 threads participate)
  auto stage = [&](int kt, __bf16* ksb, __bf16* vtb) {
    // K tile: 16KB contiguous; IOFFSET applies to both LDS and global address,
    // and the 4 chunks stride identically (4096B) on both sides.
    {
      const __bf16* g = K + (size_t)kt * 64 * 128 + t * 8;
      const unsigned l = (unsigned)(uintptr_t)(ksb + t * 8);
      asm volatile(
          "global_load_async_to_lds_b128 %0, %1, off\n\t"
          "global_load_async_to_lds_b128 %0, %1, off offset:4096\n\t"
          "global_load_async_to_lds_b128 %0, %1, off offset:8192\n\t"
          "global_load_async_to_lds_b128 %0, %1, off offset:12288"
          :: "v"(l), "v"((unsigned long long)(uintptr_t)g) : "memory");
    }
    // V^T tile: 128 rows of 64 keys; row stride differs between global (Sv)
    // and LDS (64), so compute per-chunk addresses.
#pragma unroll
    for (int j = 0; j < 4; ++j) {
      const int d   = (t >> 3) + j * 32;
      const int sub = (t & 7) * 8;
      async_b128((unsigned)(uintptr_t)(vtb + d * 64 + sub),
                 vT + (size_t)d * Sv + kt * 64 + sub);
    }
  };

  stage(0, KsBuf[0], VtBuf[0]);
  wait_async0();
  __syncthreads();

  for (int kt = 0; kt < Sv / 64; ++kt) {
    const int cur = kt & 1;
    if (kt + 1 < Sv / 64) stage(kt + 1, KsBuf[cur ^ 1], VtBuf[cur ^ 1]);
    const __bf16* Kb = KsBuf[cur];
    const __bf16* Vb = VtBuf[cur];

    // S = Q K^T  (16 rows x 64 keys per wave)
    v8f sacc[4] = {};
#pragma unroll
    for (int n = 0; n < 4; ++n) {
#pragma unroll
      for (int kk = 0; kk < 4; ++kk) {
        Frag bf = load_frag_b(Kb + (size_t)(n * 16 + ln16) * 128 + kk * 32 + (lo ? 0 : 8));
        sacc[n] = wmma_bf16(aq[kk], bf, sacc[n]);
      }
    }

    // padding mask: column (key) is lane%16 within each n-subtile
#pragma unroll
    for (int n = 0; n < 4; ++n) {
      const int keyg = kt * 64 + n * 16 + ln16;
      const float madd = mask[(size_t)b * Sv + keyg] ? 0.f : NEG_INF;
#pragma unroll
      for (int r = 0; r < 8; ++r) sacc[n][r] += madd;
    }

    // online softmax: row r+8*hi lives in VGPR r across lanes of one 16-half
    float mnew[8], sc[8];
#pragma unroll
    for (int r = 0; r < 8; ++r) {
      float mx = fmaxf(fmaxf(sacc[0][r], sacc[1][r]), fmaxf(sacc[2][r], sacc[3][r]));
#pragma unroll
      for (int off = 1; off < 16; off <<= 1) mx = fmaxf(mx, __shfl_xor(mx, off, 32));
      mnew[r] = fmaxf(mrow[r], mx);
      sc[r] = (mrow[r] == NEG_INF) ? 0.f : __expf(mrow[r] - mnew[r]);
      mrow[r] = mnew[r];
    }
#pragma unroll
    for (int r = 0; r < 8; ++r) {
      float rs = 0.f;
#pragma unroll
      for (int n = 0; n < 4; ++n) {
        float p = (mnew[r] == NEG_INF) ? 0.f : __expf(sacc[n][r] - mnew[r]);
        sacc[n][r] = p;
        rs += p;
      }
#pragma unroll
      for (int off = 1; off < 16; off <<= 1) rs += __shfl_xor(rs, off, 32);
      lrow[r] = lrow[r] * sc[r] + rs;
    }
#pragma unroll
    for (int n = 0; n < 8; ++n)
#pragma unroll
      for (int r = 0; r < 8; ++r) acc[n][r] *= sc[r];

    // P: C-layout -> LDS -> A-layout (DS ops are in-order within a wave)
#pragma unroll
    for (int n = 0; n < 4; ++n)
#pragma unroll
      for (int r = 0; r < 8; ++r)
        Ps[wave][(r + (lo ? 0 : 8)) * 64 + n * 16 + ln16] = (__bf16)sacc[n][r];
    asm volatile("" ::: "memory");

    Frag pa0 = load_frag_a(&Ps[wave][(size_t)ln16 * 64 + 0  + (lo ? 0 : 8)]);
    Frag pa1 = load_frag_a(&Ps[wave][(size_t)ln16 * 64 + 32 + (lo ? 0 : 8)]);

    // acc += P (16x64) * V (64x128), V read column-major from V^T tile
#pragma unroll
    for (int n = 0; n < 8; ++n) {
      Frag b0 = load_frag_b(Vb + (size_t)(n * 16 + ln16) * 64 + 0  + (lo ? 0 : 8));
      acc[n] = wmma_bf16(pa0, b0, acc[n]);
      Frag b1 = load_frag_b(Vb + (size_t)(n * 16 + ln16) * 64 + 32 + (lo ? 0 : 8));
      acc[n] = wmma_bf16(pa1, b1, acc[n]);
    }

    wait_async0();       // next tile's DMA has landed
    __syncthreads();     // all waves done with this buffer pair
  }

  // epilogue: normalize, store agg as bf16 (B, S, H*D)
#pragma unroll
  for (int r = 0; r < 8; ++r) {
    const float inv = lrow[r] > 0.f ? 1.f / lrow[r] : 0.f;
    const int srow = m0 + r + (lo ? 0 : 8);
#pragma unroll
    for (int n = 0; n < 8; ++n) {
      const int d = n * 16 + ln16;
      agg[((size_t)(b * Sv + srow)) * 1024 + h * 128 + d] = (__bf16)(acc[n][r] * inv);
    }
  }
}

// ---------------- output projection: (8192x1024) @ (1024x128) ----------------
__global__ __launch_bounds__(256) void out_gemm(
    const __bf16* __restrict__ aggb, const __bf16* __restrict__ woT,
    const float* __restrict__ bo, float* __restrict__ out) {
  const int lane = threadIdx.x & 31;
  const int wave = threadIdx.x >> 5;
  const bool lo  = lane < 16;
  const int ln16 = lane & 15;
  const int m0   = blockIdx.x * 128 + wave * 16;
  const int arow = m0 + ln16;

  v8f acc[8] = {};
  for (int kb = 0; kb < 1024; kb += 32) {
    Frag a = load_frag_a(aggb + (size_t)arow * 1024 + kb + (lo ? 0 : 8));
#pragma unroll
    for (int n = 0; n < 8; ++n) {
      Frag bf = load_frag_b(woT + (size_t)(n * 16 + ln16) * 1024 + kb + (lo ? 0 : 8));
      acc[n] = wmma_bf16(a, bf, acc[n]);
    }
  }
#pragma unroll
  for (int n = 0; n < 8; ++n) {
    const int col = n * 16 + ln16;
    const float bias = bo[col];
#pragma unroll
    for (int r = 0; r < 8; ++r) {
      const int row = m0 + r + (lo ? 0 : 8);
      out[(size_t)row * 128 + col] = acc[n][r] + bias;
    }
  }
}

extern "C" void kernel_launch(void* const* d_in, const int* in_sizes, int n_in,
                              void* d_out, int out_size, void* d_ws, size_t ws_size,
                              hipStream_t stream) {
  const float* x  = (const float*)d_in[0];
  const unsigned char* mask = (const unsigned char*)d_in[1];  // jnp bool -> bytes
  const float* Wq = (const float*)d_in[2];
  const float* bq = (const float*)d_in[3];
  const float* Wk = (const float*)d_in[4];
  const float* bk = (const float*)d_in[5];
  const float* Wv = (const float*)d_in[6];
  const float* bv = (const float*)d_in[7];
  const float* Wo = (const float*)d_in[8];
  const float* bo = (const float*)d_in[9];
  float* out = (float*)d_out;

  // workspace layout (bf16 elements)
  __bf16* ws  = (__bf16*)d_ws;
  __bf16* xb  = ws;                   // 1048576  x in bf16
  __bf16* wqT = xb  + 1048576;        // 131072   Wq^T (1024x128)
  __bf16* wkT = wqT + 131072;
  __bf16* wvT = wkT + 131072;
  __bf16* woT = wvT + 131072;         // Wo^T (128x1024)
  __bf16* qb  = woT + 131072;         // 8388608  q head-major (B,H,S,D)
  __bf16* kb  = qb  + 8388608;        //          k head-major (B,H,S,D)
  __bf16* vTw = kb  + 8388608;        //          v head-major transposed (B,H,D,S)
  __bf16* agg = vTw + 8388608;        // 8388608  attention output (B,S,H*D)

  cvt_bf16<<<(1048576 + 255) / 256, 256, 0, stream>>>(x, xb, 1048576);
  transpose_bf16<<<(131072 + 255) / 256, 256, 0, stream>>>(Wq, wqT, 128, 1024);
  transpose_bf16<<<(131072 + 255) / 256, 256, 0, stream>>>(Wk, wkT, 128, 1024);
  transpose_bf16<<<(131072 + 255) / 256, 256, 0, stream>>>(Wv, wvT, 128, 1024);
  transpose_bf16<<<(131072 + 255) / 256, 256, 0, stream>>>(Wo, woT, 1024, 128);

  qkv_gemm<<<dim3(64, 24), 256, 0, stream>>>(xb, wqT, wkT, wvT, bq, bk, bv, qb, kb, vTw);
  flash_attn<<<dim3(Sv / 128, Bv * Hv), 256, 0, stream>>>(qb, kb, vTw, mask, agg);
  out_gemm<<<64, 256, 0, stream>>>(agg, woT, bo, out);
}
